// DenoisingNetwork_8495445312024
// MI455X (gfx1250) — matrix operand order repaired and checked by reading, not compiled, not run on hardware
//
#include <hip/hip_runtime.h>

#define Nn 10000
#define Ee 160000
#define Hh 256
#define Kk 20
#define NODE_F 16
#define NT 10
#define ET 6
#define LL 5

typedef unsigned short ushort_t;
typedef __bf16 bf16_t;
typedef __attribute__((ext_vector_type(8)))  float  v8f;
typedef __attribute__((ext_vector_type(8)))  bf16_t v8bf;
typedef __attribute__((ext_vector_type(16))) bf16_t v16bf;

#if __has_builtin(__builtin_amdgcn_global_load_async_to_lds_b128)
#define ASYNC_LDS 1
typedef int v4i_vs __attribute__((vector_size(16)));
typedef __attribute__((address_space(1))) v4i_vs* gas_p;
typedef __attribute__((address_space(3))) v4i_vs* las_p;
#endif

__device__ __forceinline__ ushort_t f2bf(float f) {
  union { float f; unsigned u; } v; v.f = f;
  unsigned r = v.u + 0x7FFFu + ((v.u >> 16) & 1u);
  return (ushort_t)(r >> 16);
}

// 16-byte global -> LDS stage; async DMA on CDNA5 (ASYNCcnt), sync fallback otherwise
__device__ __forceinline__ void stage16B(ushort_t* l, const ushort_t* g) {
#ifdef ASYNC_LDS
  __builtin_amdgcn_global_load_async_to_lds_b128(
      (gas_p)(unsigned long long)g,
      (las_p)(unsigned)(unsigned long long)l, 0, 0);
#else
  *(uint4*)l = *(const uint4*)g;
#endif
}
__device__ __forceinline__ void stage_wait() {
#ifdef ASYNC_LDS
  asm volatile("s_wait_asynccnt 0" ::: "memory");
#endif
}

__device__ __forceinline__ v16bf frag16(const ushort_t* p0, const ushort_t* p1) {
  v8bf lo = *(const v8bf*)p0;
  v8bf hi = *(const v8bf*)p1;
  return __builtin_shufflevector(lo, hi, 0,1,2,3,4,5,6,7,8,9,10,11,12,13,14,15);
}
// A fragment (16x32 bf16): lanes 0-15 hold K {0..7,16..23}, lanes 16-31 hold K {8..15,24..31}
__device__ __forceinline__ v16bf fragA(const ushort_t* row, int kk, int laneHi) {
  const ushort_t* p = row + kk + 8 * laneHi;
  return frag16(p, p + 16);
}
// B fragment (32x16 bf16): lane = column; lanes 0-15 hold K {0..7,16..23}, lanes 16-31 hold K {8..15,24..31}
__device__ __forceinline__ v16bf fragB(const ushort_t* col, int kk, int laneHi) {
  const ushort_t* p = col + kk + 8 * laneHi;
  return frag16(p, p + 16);
}
__device__ __forceinline__ v8f wmma_bf(v16bf a, v16bf b, v8f c) {
  return __builtin_amdgcn_wmma_f32_16x16x32_bf16(false, a, false, b, (short)0, c, false, false);
}
__device__ __forceinline__ float sigmoidf_(float x) { return 1.0f / (1.0f + __expf(-x)); }

/* ---------------- small utility kernels ---------------- */

__global__ void k_zero_f32(float* p, int n) {
  int i = blockIdx.x * 256 + threadIdx.x;
  if (i < n) p[i] = 0.0f;
}
__global__ void k_cvt_bf(ushort_t* dst, const float* src, int n) {
  int i = blockIdx.x * 256 + threadIdx.x;
  if (i < n) dst[i] = f2bf(src[i]);
}
// dst[c*fi+k] = bf16(src[k*fo+c])  (transpose + convert)
__global__ void k_packT_bf(ushort_t* dst, const float* src, int fi, int fo) {
  int i = blockIdx.x * 256 + threadIdx.x;
  if (i >= fi * fo) return;
  int k = i % fi, c = i / fi;
  dst[i] = f2bf(src[k * fo + c]);
}
// Whh (768x256) -> WhhT (256x768) f32
__global__ void k_transpose_whh(float* dst, const float* src) {
  int i = blockIdx.x * 256 + threadIdx.x;     // over 256*768
  int j = i % 768, k = i / 768;
  dst[i] = src[j * 256 + k];
}

/* ---------------- embeddings ---------------- */

__global__ __launch_bounds__(256) void k_node_emb(const float* __restrict__ x,
                                                  const float* __restrict__ w,
                                                  const float* __restrict__ b,
                                                  float* __restrict__ hv,
                                                  ushort_t* __restrict__ hvbf) {
  int n = blockIdx.x, c = threadIdx.x;
  float acc = b[c];
#pragma unroll
  for (int k = 0; k < NODE_F; ++k) acc += x[n * NODE_F + k] * w[k * Hh + c];
  hv[n * Hh + c] = acc;
  hvbf[n * Hh + c] = f2bf(acc);
}
__global__ __launch_bounds__(256) void k_edge_emb(const float* __restrict__ ea,
                                                  const float* __restrict__ w,
                                                  const float* __restrict__ b,
                                                  ushort_t* __restrict__ hebf) {
  int e = blockIdx.x, c = threadIdx.x;
  hebf[(size_t)e * Hh + c] = f2bf(ea[e] * w[c] + b[c]);
}

/* ------- GEMM 1: gathered feat(E x 768) @ [Wf1|Wg1] -> relu -> hid(E x 512) bf16 ------- */

__global__ __launch_bounds__(256) void k_gemm1(const ushort_t* __restrict__ hvbf,
                                               const ushort_t* __restrict__ hebf,
                                               const int* __restrict__ eidx,
                                               const ushort_t* __restrict__ w1p,   // 512 x 768
                                               const float* __restrict__ bf1,
                                               const float* __restrict__ bg1,
                                               ushort_t* __restrict__ hid) {
  __shared__ ushort_t As[2][64 * 32];
  const int tid = threadIdx.x;
  const int lane = tid & 31, wave = tid >> 5;
  const int eBase = blockIdx.x * 64;
  const int nBase = blockIdx.y * 64;
  const int mSub = wave & 3, nPair = wave >> 2;
  const int laneLo = lane & 15, laneHi = lane >> 4;
  const int* srcRow = eidx;
  const int* dstRow = eidx + Ee;

  const int aRow = tid >> 2, aChunk = tid & 3;
  const int aEdge = eBase + aRow;

  const int col0 = nBase + nPair * 32 + laneLo;
  const int col1 = col0 + 16;
  const ushort_t* b0base = w1p + (size_t)col0 * 768;
  const ushort_t* b1base = w1p + (size_t)col1 * 768;

  auto stageA = [&](int kk, int buf) {
    int kg = kk + aChunk * 8;
    int seg = kg >> 8, off = kg & 255;
    const ushort_t* sp;
    if (seg == 0)      sp = hvbf + (size_t)dstRow[aEdge] * Hh + off;
    else if (seg == 1) sp = hvbf + (size_t)srcRow[aEdge] * Hh + off;
    else               sp = hebf + (size_t)aEdge * Hh + off;
    stage16B(&As[buf][aRow * 32 + aChunk * 8], sp);
  };

  stageA(0, 0);
  v8f acc0 = {}, acc1 = {};
  for (int i = 0; i < 24; ++i) {
    const int kk = i * 32;
    const int cur = i & 1;
    stage_wait();
    __syncthreads();
    if (i + 1 < 24) stageA(kk + 32, cur ^ 1);
    const ushort_t* arow = &As[cur][(mSub * 16 + laneLo) * 32];
    v16bf a  = fragA(arow, 0, laneHi);
    v16bf b0 = fragB(b0base, kk, laneHi);
    v16bf b1 = fragB(b1base, kk, laneHi);
    acc0 = wmma_bf(a, b0, acc0);
    acc1 = wmma_bf(a, b1, acc1);
  }
  float bias0 = (col0 < 256) ? bf1[col0] : bg1[col0 - 256];
  float bias1 = (col1 < 256) ? bf1[col1] : bg1[col1 - 256];
#pragma unroll
  for (int r = 0; r < 8; ++r) {
    int e = eBase + mSub * 16 + r + 8 * laneHi;
    hid[(size_t)e * 512 + col0] = f2bf(fmaxf(acc0[r] + bias0, 0.0f));
    hid[(size_t)e * 512 + col1] = f2bf(fmaxf(acc1[r] + bias1, 0.0f));
  }
}

/* ------- GEMM 2: m = (hidF@Wf2+b)*(hidG@Wg2+b); scatter-add to agg[dst] ------- */

__global__ __launch_bounds__(256) void k_gemm2(const ushort_t* __restrict__ hid,
                                               const ushort_t* __restrict__ w2p,  // 512 x 256
                                               const float* __restrict__ bf2,
                                               const float* __restrict__ bg2,
                                               const int* __restrict__ eidx,
                                               float* __restrict__ agg) {
  __shared__ ushort_t As[2][64 * 64];   // per K-step: 64 edges x (32 f | 32 g)
  const int tid = threadIdx.x;
  const int lane = tid & 31, wave = tid >> 5;
  const int eBase = blockIdx.x * 64;
  const int nBase = blockIdx.y * 64;
  const int mSub = wave & 3, nPair = wave >> 2;
  const int laneLo = lane & 15, laneHi = lane >> 4;
  const int* dstRow = eidx + Ee;

  const int aRow = tid >> 2, aChunk = tid & 3;
  const ushort_t* aBase = hid + (size_t)(eBase + aRow) * 512 + aChunk * 8;

  const int col0 = nBase + nPair * 32 + laneLo;
  const int col1 = col0 + 16;
  const ushort_t* bF0 = w2p + (size_t)col0 * 256;
  const ushort_t* bF1 = w2p + (size_t)col1 * 256;
  const ushort_t* bG0 = w2p + (size_t)(256 + col0) * 256;
  const ushort_t* bG1 = w2p + (size_t)(256 + col1) * 256;

  auto stageA = [&](int kk, int buf) {
    stage16B(&As[buf][aRow * 64 + aChunk * 8],      aBase + kk);        // f half
    stage16B(&As[buf][aRow * 64 + 32 + aChunk * 8], aBase + 256 + kk);  // g half
  };

  stageA(0, 0);
  v8f aF0 = {}, aF1 = {}, aG0 = {}, aG1 = {};
  for (int i = 0; i < 8; ++i) {
    const int kk = i * 32;
    const int cur = i & 1;
    stage_wait();
    __syncthreads();
    if (i + 1 < 8) stageA(kk + 32, cur ^ 1);
    const ushort_t* rf = &As[cur][(mSub * 16 + laneLo) * 64];
    v16bf af = fragA(rf, 0, laneHi);
    v16bf ag = fragA(rf + 32, 0, laneHi);
    v16bf b;
    b = fragB(bF0, kk, laneHi); aF0 = wmma_bf(af, b, aF0);
    b = fragB(bF1, kk, laneHi); aF1 = wmma_bf(af, b, aF1);
    b = fragB(bG0, kk, laneHi); aG0 = wmma_bf(ag, b, aG0);
    b = fragB(bG1, kk, laneHi); aG1 = wmma_bf(ag, b, aG1);
  }
  float cF0 = bf2[col0], cF1 = bf2[col1], cG0 = bg2[col0], cG1 = bg2[col1];
#pragma unroll
  for (int r = 0; r < 8; ++r) {
    int e = eBase + mSub * 16 + r + 8 * laneHi;
    int d = dstRow[e];
    atomicAdd(&agg[(size_t)d * Hh + col0], (aF0[r] + cF0) * (aG0[r] + cG0));
    atomicAdd(&agg[(size_t)d * Hh + col1], (aF1[r] + cF1) * (aG1[r] + cG1));
  }
}

/* ------- GEMM 3: gi = [h_v | agg](N x 512) @ Wih^T + bih -> (N x 768) f32 ------- */

__global__ __launch_bounds__(256) void k_gemm3(const ushort_t* __restrict__ hvbf,
                                               const ushort_t* __restrict__ aggbf,
                                               const ushort_t* __restrict__ wihbf, // 768 x 512
                                               const float* __restrict__ bih,
                                               float* __restrict__ gi) {
  __shared__ ushort_t As[2][64 * 32];
  const int tid = threadIdx.x;
  const int lane = tid & 31, wave = tid >> 5;
  const int rBase = blockIdx.x * 64;
  const int cBase = blockIdx.y * 64;
  const int mSub = wave & 3, nPair = wave >> 2;
  const int laneLo = lane & 15, laneHi = lane >> 4;

  const int aRow = tid >> 2, aChunk = tid & 3;
  int aNode = rBase + aRow; if (aNode >= Nn) aNode = Nn - 1;

  const int col0 = cBase + nPair * 32 + laneLo;
  const int col1 = col0 + 16;
  const ushort_t* b0base = wihbf + (size_t)col0 * 512;
  const ushort_t* b1base = wihbf + (size_t)col1 * 512;

  auto stageA = [&](int kk, int buf) {
    int kg = kk + aChunk * 8;
    int seg = kg >> 8, off = kg & 255;
    const ushort_t* sp = (seg == 0 ? hvbf : aggbf) + (size_t)aNode * Hh + off;
    stage16B(&As[buf][aRow * 32 + aChunk * 8], sp);
  };

  stageA(0, 0);
  v8f acc0 = {}, acc1 = {};
  for (int i = 0; i < 16; ++i) {
    const int kk = i * 32;
    const int cur = i & 1;
    stage_wait();
    __syncthreads();
    if (i + 1 < 16) stageA(kk + 32, cur ^ 1);
    const ushort_t* arow = &As[cur][(mSub * 16 + laneLo) * 32];
    v16bf a  = fragA(arow, 0, laneHi);
    v16bf b0 = fragB(b0base, kk, laneHi);
    v16bf b1 = fragB(b1base, kk, laneHi);
    acc0 = wmma_bf(a, b0, acc0);
    acc1 = wmma_bf(a, b1, acc1);
  }
  float c0 = bih[col0], c1 = bih[col1];
#pragma unroll
  for (int r = 0; r < 8; ++r) {
    int n = rBase + mSub * 16 + r + 8 * laneHi;
    if (n < Nn) {
      gi[(size_t)n * 768 + col0] = acc0[r] + c0;
      gi[(size_t)n * 768 + col1] = acc1[r] + c1;
    }
  }
}

/* ------- sequential GRU scan over nodes (1 workgroup, 768 threads) ------- */

__global__ __launch_bounds__(768) void k_gru_scan(const float* __restrict__ gi,
                                                  const float* __restrict__ whhT,  // 256 x 768
                                                  const float* __restrict__ bhh,
                                                  float* __restrict__ hv,
                                                  ushort_t* __restrict__ hvbf) {
  __shared__ float h[Hh];
  __shared__ float gh[3 * Hh];
  const int t = threadIdx.x;
  if (t < Hh) h[t] = 0.0f;
  const float bh = bhh[t];
  __syncthreads();
  for (int n = 0; n < Nn; ++n) {
    float acc = bh;
#pragma unroll 8
    for (int k = 0; k < Hh; ++k) acc += h[k] * whhT[k * 768 + t];
    gh[t] = acc;
    __syncthreads();
    if (t < Hh) {
      const float* g = gi + (size_t)n * 768;
      float r = sigmoidf_(g[t] + gh[t]);
      float z = sigmoidf_(g[Hh + t] + gh[Hh + t]);
      float nn = tanhf(g[2 * Hh + t] + r * gh[2 * Hh + t]);
      float hn = (1.0f - z) * nn + z * h[t];
      h[t] = hn;
      hv[(size_t)n * Hh + t] = hn;
      hvbf[(size_t)n * Hh + t] = f2bf(hn);
    }
    __syncthreads();
  }
}

/* ---------------- readout ---------------- */

__global__ __launch_bounds__(256) void k_mean_g(const float* __restrict__ hv, float* __restrict__ g) {
  int c = threadIdx.x;
  float acc = 0.0f;
  for (int n = 0; n < Nn; ++n) acc += hv[(size_t)n * Hh + c];
  g[c] = acc / (float)Nn;
}

__global__ __launch_bounds__(256) void k_prep(const float* __restrict__ g,
                                              const float* __restrict__ hv,
                                              const float* __restrict__ Wnp1, const float* __restrict__ bnp1,
                                              const float* __restrict__ Wa1,  const float* __restrict__ ba1,
                                              float* __restrict__ gpn, float* __restrict__ gpa,
                                              float* __restrict__ accN, float* __restrict__ accA) {
  __shared__ float gs[Hh], hl[Hh];
  int c = threadIdx.x;
  gs[c] = g[c];
  hl[c] = hv[(size_t)(Nn - 1) * Hh + c];
  __syncthreads();
  float aN = bnp1[c], aA = ba1[c];
  for (int k = 0; k < Hh; ++k) {
    aN += gs[k] * Wnp1[k * Hh + c];
    aA += gs[k] * Wa1[k * Hh + c] + hl[k] * Wa1[(Hh + k) * Hh + c];
  }
  gpn[c] = aN;
  gpa[c] = aA;
  if (c < NT) accN[c] = 0.0f;
  if (c < Kk) accA[c] = 0.0f;
}

__global__ __launch_bounds__(256) void k_node_alpha(const float* __restrict__ hv,
                                                    const float* __restrict__ gpn,
                                                    const float* __restrict__ gpa,
                                                    const float* __restrict__ Wnp1,
                                                    const float* __restrict__ Wnp2, const float* __restrict__ bnp2,
                                                    const float* __restrict__ Wa1,
                                                    const float* __restrict__ Wa2,  const float* __restrict__ ba2,
                                                    float* __restrict__ accN, float* __restrict__ accA) {
  __shared__ float hvn[Hh], hid[Hh];
  int n = blockIdx.x, c = threadIdx.x;
  hvn[c] = hv[(size_t)n * Hh + c];
  __syncthreads();
  float a = gpn[c];
  for (int k = 0; k < Hh; ++k) a += hvn[k] * Wnp1[(Hh + k) * Hh + c];
  hid[c] = fmaxf(a, 0.0f);
  __syncthreads();
  if (c < NT) {
    float o = bnp2[c];
    for (int j = 0; j < Hh; ++j) o += hid[j] * Wnp2[j * NT + c];
    atomicAdd(&accN[c], o);
  }
  __syncthreads();
  a = gpa[c];
  for (int k = 0; k < Hh; ++k) a += hvn[k] * Wa1[(2 * Hh + k) * Hh + c];
  hid[c] = fmaxf(a, 0.0f);
  __syncthreads();
  if (c < Kk) {
    float o = ba2[c];
    for (int j = 0; j < Hh; ++j) o += hid[j] * Wa2[j * Kk + c];
    atomicAdd(&accA[c], o);
  }
}

__global__ void k_finalize(const float* __restrict__ accN, const float* __restrict__ accA,
                           float* __restrict__ out, float* __restrict__ alph) {
  if (threadIdx.x != 0) return;
  float v[Kk];
  float m = -1e30f;
  for (int o = 0; o < NT; ++o) { v[o] = accN[o] / (float)Nn; m = fmaxf(m, v[o]); }
  float s = 0.0f;
  for (int o = 0; o < NT; ++o) { v[o] = __expf(v[o] - m); s += v[o]; }
  for (int o = 0; o < NT; ++o) out[o] = v[o] / s;
  m = -1e30f;
  for (int o = 0; o < Kk; ++o) { v[o] = accA[o]; m = fmaxf(m, v[o]); }
  s = 0.0f;
  for (int o = 0; o < Kk; ++o) { v[o] = __expf(v[o] - m); s += v[o]; }
  for (int o = 0; o < Kk; ++o) alph[o] = v[o] / s;
}

__global__ __launch_bounds__(256) void k_edge_pred(const float* __restrict__ hv,
                                                   const float* __restrict__ We1, const float* __restrict__ be1,
                                                   const float* __restrict__ We2, const float* __restrict__ be2,
                                                   const float* __restrict__ alph,
                                                   float* __restrict__ out) {
  __shared__ float hvn[Hh], hid[Hh], lg[ET * Kk], pm[ET * Kk];
  int n = blockIdx.x, c = threadIdx.x;
  hvn[c] = hv[(size_t)n * Hh + c];
  __syncthreads();
  float a = be1[c];
  for (int k = 0; k < Hh; ++k) a += hvn[k] * We1[k * Hh + c];
  hid[c] = fmaxf(a, 0.0f);
  __syncthreads();
  if (c < ET * Kk) {
    float o = be2[c];
    for (int j = 0; j < Hh; ++j) o += hid[j] * We2[j * ET * Kk + c];
    lg[c] = o;
  }
  __syncthreads();
  if (c < Kk) {
    float m = -1e30f;
    for (int t = 0; t < ET; ++t) m = fmaxf(m, lg[t * Kk + c]);
    float ex[ET]; float s = 0.0f;
    for (int t = 0; t < ET; ++t) { ex[t] = __expf(lg[t * Kk + c] - m); s += ex[t]; }
    float al = alph[c] / s;
    for (int t = 0; t < ET; ++t) pm[t * Kk + c] = al * ex[t];
  }
  __syncthreads();
  if (c < ET) {
    float p = 0.0f;
    for (int k = 0; k < Kk; ++k) p += pm[c * Kk + k];
    out[NT + (size_t)n * ET + c] = p;
  }
}

/* ---------------- host orchestration ---------------- */

extern "C" void kernel_launch(void* const* d_in, const int* in_sizes, int n_in,
                              void* d_out, int out_size, void* d_ws, size_t ws_size,
                              hipStream_t stream) {
  const float* x     = (const float*)d_in[0];
  const int*   eidx  = (const int*)d_in[1];
  const float* eattr = (const float*)d_in[2];
  const float* Wne   = (const float*)d_in[3];
  const float* bne   = (const float*)d_in[4];
  const float* Wee   = (const float*)d_in[5];
  const float* bee   = (const float*)d_in[6];

  char* p = (char*)d_ws;
  auto carve = [&](size_t bytes) -> char* {
    char* r = p; p += (bytes + 255) & ~(size_t)255; return r;
  };
  float*    hv    = (float*)   carve((size_t)Nn * Hh * 4);
  ushort_t* hvbf  = (ushort_t*)carve((size_t)Nn * Hh * 2);
  ushort_t* hebf  = (ushort_t*)carve((size_t)Ee * Hh * 2);
  ushort_t* hid   = (ushort_t*)carve((size_t)Ee * 512 * 2);
  float*    agg   = (float*)   carve((size_t)Nn * Hh * 4);
  ushort_t* aggbf = (ushort_t*)carve((size_t)Nn * Hh * 2);
  float*    gi    = (float*)   carve((size_t)Nn * 768 * 4);
  ushort_t* w1p   = (ushort_t*)carve((size_t)512 * 768 * 2);
  ushort_t* w2p   = (ushort_t*)carve((size_t)512 * 256 * 2);
  ushort_t* wihbf = (ushort_t*)carve((size_t)768 * 512 * 2);
  float*    whhT  = (float*)   carve((size_t)256 * 768 * 4);
  float*    gbuf  = (float*)   carve(Hh * 4);
  float*    gpn   = (float*)   carve(Hh * 4);
  float*    gpa   = (float*)   carve(Hh * 4);
  float*    accN  = (float*)   carve(NT * 4);
  float*    accA  = (float*)   carve(Kk * 4);
  float*    alph  = (float*)   carve(Kk * 4);
  float*    out   = (float*)d_out;

  k_node_emb<<<Nn, 256, 0, stream>>>(x, Wne, bne, hv, hvbf);
  k_edge_emb<<<Ee, 256, 0, stream>>>(eattr, Wee, bee, hebf);

  for (int l = 0; l < LL; ++l) {
    int b = 7 + 12 * l;
    const float* Wf1 = (const float*)d_in[b + 0];
    const float* bf1 = (const float*)d_in[b + 1];
    const float* Wf2 = (const float*)d_in[b + 2];
    const float* bf2 = (const float*)d_in[b + 3];
    const float* Wg1 = (const float*)d_in[b + 4];
    const float* bg1 = (const float*)d_in[b + 5];
    const float* Wg2 = (const float*)d_in[b + 6];
    const float* bg2 = (const float*)d_in[b + 7];
    const float* Wih = (const float*)d_in[b + 8];
    const float* Whh = (const float*)d_in[b + 9];
    const float* bih = (const float*)d_in[b + 10];
    const float* bhh = (const float*)d_in[b + 11];

    k_packT_bf<<<(768 * 256 + 255) / 256, 256, 0, stream>>>(w1p, Wf1, 768, 256);
    k_packT_bf<<<(768 * 256 + 255) / 256, 256, 0, stream>>>(w1p + (size_t)256 * 768, Wg1, 768, 256);
    k_packT_bf<<<(256 * 256 + 255) / 256, 256, 0, stream>>>(w2p, Wf2, 256, 256);
    k_packT_bf<<<(256 * 256 + 255) / 256, 256, 0, stream>>>(w2p + (size_t)256 * 256, Wg2, 256, 256);
    k_cvt_bf<<<(768 * 512 + 255) / 256, 256, 0, stream>>>(wihbf, Wih, 768 * 512);
    k_transpose_whh<<<(256 * 768 + 255) / 256, 256, 0, stream>>>(whhT, Whh);
    k_zero_f32<<<(Nn * Hh + 255) / 256, 256, 0, stream>>>(agg, Nn * Hh);

    k_gemm1<<<dim3(Ee / 64, 8), 256, 0, stream>>>(hvbf, hebf, eidx, w1p, bf1, bg1, hid);
    k_gemm2<<<dim3(Ee / 64, 4), 256, 0, stream>>>(hid, w2p, bf2, bg2, eidx, agg);
    k_cvt_bf<<<(Nn * Hh + 255) / 256, 256, 0, stream>>>(aggbf, agg, Nn * Hh);
    k_gemm3<<<dim3((Nn + 63) / 64, 12), 256, 0, stream>>>(hvbf, aggbf, wihbf, bih, gi);
    k_gru_scan<<<1, 768, 0, stream>>>(gi, whhT, bhh, hv, hvbf);
  }

  const float* Wa1  = (const float*)d_in[67];
  const float* ba1  = (const float*)d_in[68];
  const float* Wa2  = (const float*)d_in[69];
  const float* ba2  = (const float*)d_in[70];
  const float* Wnp1 = (const float*)d_in[71];
  const float* bnp1 = (const float*)d_in[72];
  const float* Wnp2 = (const float*)d_in[73];
  const float* bnp2 = (const float*)d_in[74];
  const float* We1  = (const float*)d_in[75];
  const float* be1  = (const float*)d_in[76];
  const float* We2  = (const float*)d_in[77];
  const float* be2  = (const float*)d_in[78];

  k_mean_g<<<1, 256, 0, stream>>>(hv, gbuf);
  k_prep<<<1, 256, 0, stream>>>(gbuf, hv, Wnp1, bnp1, Wa1, ba1, gpn, gpa, accN, accA);
  k_node_alpha<<<Nn, 256, 0, stream>>>(hv, gpn, gpa, Wnp1, Wnp2, bnp2, Wa1, Wa2, ba2, accN, accA);
  k_finalize<<<1, 32, 0, stream>>>(accN, accA, out, alph);
  k_edge_pred<<<Nn, 256, 0, stream>>>(hv, We1, be1, We2, be2, alph, out);
  (void)in_sizes; (void)n_in; (void)out_size; (void)ws_size;
}
